// TraceClassifier_12644383719725
// MI455X (gfx1250) — compile-verified, matrix operand order checked by reading, not run on hardware
//
#include <hip/hip_runtime.h>
#include <math.h>

// ---------------------------------------------------------------------------
// CDNA5 (gfx1250) WMMA helpers: D = A(16x32 bf16) x B(32x16 bf16) + C(16x16 f32)
// ---------------------------------------------------------------------------
typedef __attribute__((ext_vector_type(8)))  float  v8f;
typedef __attribute__((ext_vector_type(16))) __bf16 v16bf;

union AFrag { v16bf v; unsigned short u[16]; uint4 q[2]; };

__device__ __forceinline__ unsigned short f2bf(float f) {
    unsigned u = __float_as_uint(f);
    u += 0x7FFFu + ((u >> 16) & 1u);   // round-to-nearest-even
    return (unsigned short)(u >> 16);
}
__device__ __forceinline__ float sigm(float x) { return 1.0f / (1.0f + __expf(-x)); }

__device__ __forceinline__ v8f wmma_bf16(v16bf a, v16bf b, v8f c) {
    // 8 args: (neg_a, A, neg_b, B, c_mod, C, reuse_a, reuse_b)
    return __builtin_amdgcn_wmma_f32_16x16x32_bf16(false, a, false, b, (short)0, c, false, false);
}

#define GW 4           // waves (16-row tiles) per block; blockDim = 128
#define LEVELS 8       // >= max tree height + 1 (actual = 6); extras are exact no-ops

// Element K-index inside a 16-bit A (16x32) / B (32x16) fragment:
//   K(j, hi) = 16*(j>>3) + 8*hi + (j&7),   hi = lane>>4, row/col = lane&15
// (ISA 7.12.2: lanes 0-15 VGPR0..3 hold K=0..7, VGPR4..7 K=16..23; lanes 16-31 +8)

// ---------------------------------------------------------------------------
// Weight -> bf16 WMMA-B fragment layout.  W is row-major [M,K], out holds
// (M/16)*(K/32) fragments of 512 bf16:  out[((nt*(K/32)+ks)*32+lane)*16+j]
//   = bf16( W[(nt*16 + (lane&15))*K + ks*32 + Kidx(j,lane>>4)] )
// ---------------------------------------------------------------------------
__global__ void wfrag_kernel(const float* __restrict__ W, unsigned short* __restrict__ out,
                             int M, int K) {
    int t = blockIdx.x * blockDim.x + threadIdx.x;
    if (t >= M * K) return;
    int j    = t & 15;
    int lane = (t >> 4) & 31;
    int f    = t >> 9;
    int ksteps = K >> 5;
    int nt = f / ksteps;
    int ks = f - nt * ksteps;
    int r = lane & 15, hi = lane >> 4;
    int kj = 16 * (j >> 3) + 8 * hi + (j & 7);
    out[t] = f2bf(W[(nt * 16 + r) * K + ks * 32 + kj]);
}

// ---------------------------------------------------------------------------
// Generic WMMA GEMM:  out[N,M] = act( (optional rowscale * maybe-relu(A))[N,K]
//                                     @ W[M,K]^T + bias )
// flags: bit0 = relu on input rows, bit1 = relu on output.
// In-place (out==A) is safe: each block stages its own 64 rows before writing.
// ---------------------------------------------------------------------------
__global__ void gemm_wmma_kernel(const float* __restrict__ A, const float* __restrict__ rowscale,
                                 const unsigned short* __restrict__ Wfrag,
                                 const float* __restrict__ bias,
                                 float* __restrict__ out, int Nn, int K, int M, int flags) {
    __shared__ unsigned short lds[GW][16][256];
    int lane = threadIdx.x & 31, w = threadIdx.x >> 5;
    int tbase = (blockIdx.x * GW + w) * 16;
    for (int idx = lane; idx < 16 * K; idx += 32) {
        int rr = idx / K, kk = idx - rr * K;
        int row = tbase + rr;
        float v = 0.f;
        if (row < Nn) {
            v = A[(long)row * K + kk];
            if (flags & 1) v = fmaxf(v, 0.f);
            if (rowscale)  v *= rowscale[row];
        }
        lds[w][rr][kk] = f2bf(v);
    }
    __syncthreads();
    if (tbase >= Nn) return;
    int r = lane & 15, hi = lane >> 4;
    int ksteps = K >> 5, ntiles = M >> 4;
    for (int nt = 0; nt < ntiles; ++nt) {
        v8f acc = {0.f,0.f,0.f,0.f,0.f,0.f,0.f,0.f};
        for (int ks = 0; ks < ksteps; ++ks) {
            AFrag a, b;
            const uint4* pa = (const uint4*)&lds[w][r][ks * 32 + 8 * hi];
            a.q[0] = pa[0]; a.q[1] = pa[2];
            const uint4* pb = (const uint4*)&Wfrag[((nt * ksteps + ks) * 32 + lane) * 16];
            b.q[0] = pb[0]; b.q[1] = pb[1];
            acc = wmma_bf16(a.v, b.v, acc);
        }
        int col = nt * 16 + r;
        float bv = bias ? bias[col] : 0.f;
        #pragma unroll
        for (int v = 0; v < 8; ++v) {
            int row = tbase + v + 8 * hi;
            if (row < Nn) {
                float val = acc[v] + bv;
                if (flags & 2) val = fmaxf(val, 0.f);
                out[(long)row * M + col] = val;
            }
        }
    }
}

// ---------------------------------------------------------------------------
// Fused embedding gather + latency MLP + merge GEMM (K=192, M=64, ReLU out)
// ---------------------------------------------------------------------------
__global__ void embed_merge_kernel(
    const int* __restrict__ api_id, const int* __restrict__ status_id,
    const int* __restrict__ node_id, const int* __restrict__ depth,
    const int* __restrict__ pos, const float* __restrict__ lat,
    const float* __restrict__ api_emb, const float* __restrict__ status_emb,
    const float* __restrict__ node_emb, const float* __restrict__ depth_emb,
    const float* __restrict__ pos_emb,
    const float* __restrict__ latW1, const float* __restrict__ latb1,
    const float* __restrict__ latW2, const float* __restrict__ latb2,
    const unsigned short* __restrict__ mergeWfrag, const float* __restrict__ merge_b,
    float* __restrict__ xout, int Nn) {
    __shared__ unsigned short lds[GW][16][192];
    int lane = threadIdx.x & 31, w = threadIdx.x >> 5;
    int tbase = (blockIdx.x * GW + w) * 16;
    int r = lane & 15, hi = lane >> 4;
    int node = tbase + r;
    bool valid = node < Nn;
    if (hi == 0) {
        int a  = valid ? api_id[node]    : 0;
        int s  = valid ? status_id[node] : 0;
        int nd = valid ? node_id[node]   : 0;
        for (int k = 0; k < 32; ++k) {
            lds[w][r][k]      = f2bf(valid ? api_emb[a * 32 + k]    : 0.f);
            lds[w][r][32 + k] = f2bf(valid ? status_emb[s * 32 + k] : 0.f);
            lds[w][r][64 + k] = f2bf(valid ? node_emb[nd * 32 + k]  : 0.f);
        }
    } else {
        int d = valid ? depth[node] : 0; d = min(max(d, 0), 63);
        int p = valid ? pos[node]   : 0; p = min(max(p, 0), 511);
        float lv = valid ? lat[node] : 0.f;
        float v1[32];
        #pragma unroll
        for (int j = 0; j < 32; ++j) v1[j] = fmaxf(lv * latW1[j] + latb1[j], 0.f);
        for (int k = 0; k < 32; ++k) {
            lds[w][r][96 + k]  = f2bf(valid ? depth_emb[d * 32 + k] : 0.f);
            lds[w][r][128 + k] = f2bf(valid ? pos_emb[p * 32 + k]   : 0.f);
            float lh = latb2[k];
            #pragma unroll
            for (int j = 0; j < 32; ++j) lh += v1[j] * latW2[k * 32 + j];
            lds[w][r][160 + k] = f2bf(valid ? lh : 0.f);
        }
    }
    __syncthreads();
    if (tbase >= Nn) return;
    for (int nt = 0; nt < 4; ++nt) {
        v8f acc = {0.f,0.f,0.f,0.f,0.f,0.f,0.f,0.f};
        for (int ks = 0; ks < 6; ++ks) {
            AFrag a, b;
            const uint4* pa = (const uint4*)&lds[w][r][ks * 32 + 8 * hi];
            a.q[0] = pa[0]; a.q[1] = pa[2];
            const uint4* pb = (const uint4*)&mergeWfrag[((nt * 6 + ks) * 32 + lane) * 16];
            b.q[0] = pb[0]; b.q[1] = pb[1];
            acc = wmma_bf16(a.v, b.v, acc);
        }
        int col = nt * 16 + r;
        float bv = merge_b[col];
        #pragma unroll
        for (int v = 0; v < 8; ++v) {
            int row = tbase + v + 8 * hi;
            if (row < Nn) xout[(long)row * 64 + col] = fmaxf(acc[v] + bv, 0.f);
        }
    }
}

// ---------------------------------------------------------------------------
// Utility / graph kernels
// ---------------------------------------------------------------------------
__global__ void fill_kernel(float* __restrict__ p, float v, long n) {
    long i = (long)blockIdx.x * blockDim.x + threadIdx.x;
    if (i < n) p[i] = v;
}
__global__ void call_deg_kernel(const int* __restrict__ src, const int* __restrict__ dst,
                                float* __restrict__ deg, int E) {
    int e = blockIdx.x * blockDim.x + threadIdx.x;
    if (e >= E) return;
    atomicAdd(&deg[src[e]], 1.f);
    atomicAdd(&deg[dst[e]], 1.f);
}
__global__ void host_deg_kernel(const int* __restrict__ hs, const int* __restrict__ hd,
                                float* __restrict__ dout, float* __restrict__ din, int E) {
    int e = blockIdx.x * blockDim.x + threadIdx.x;
    if (e >= E) return;
    atomicAdd(&dout[hs[e]], 1.f);
    atomicAdd(&din[hd[e]], 1.f);
}
__global__ void rsqrt_kernel(float* __restrict__ p, int n) {
    int i = blockIdx.x * blockDim.x + threadIdx.x;
    if (i < n) p[i] = rsqrtf(fmaxf(p[i], 1.f));
}
// agg[n,:] = x[n,:]*s[n]  (self-loop contribution, source-scaled)
__global__ void initagg_kernel(const float* __restrict__ x, const float* __restrict__ s,
                               float* __restrict__ agg, int Nn) {
    long i = (long)blockIdx.x * blockDim.x + threadIdx.x;
    if (i >= (long)Nn * 64) return;
    int row = (int)(i >> 6);
    agg[i] = x[i] * s[row];
}
// symmetrized call-graph scatter: both directions per listed edge
__global__ void scatter_call_kernel(const float* __restrict__ x, const float* __restrict__ s,
                                    const int* __restrict__ src, const int* __restrict__ dst,
                                    float* __restrict__ agg, int E) {
    int t = blockIdx.x * blockDim.x + threadIdx.x;
    int e = t >> 2;
    if (e >= E) return;
    int c0 = (t & 3) * 16;
    int a = src[e], b = dst[e];
    float sa = s[a], sb = s[b];
    #pragma unroll
    for (int c = 0; c < 16; ++c) {
        atomicAdd(&agg[(long)b * 64 + c0 + c], x[(long)a * 64 + c0 + c] * sa);
        atomicAdd(&agg[(long)a * 64 + c0 + c], x[(long)b * 64 + c0 + c] * sb);
    }
}
__global__ void scatter_host_kernel(const float* __restrict__ x, const float* __restrict__ s,
                                    const int* __restrict__ hs, const int* __restrict__ hd,
                                    float* __restrict__ agg, int E) {
    int t = blockIdx.x * blockDim.x + threadIdx.x;
    int e = t >> 2;
    if (e >= E) return;
    int c0 = (t & 3) * 16;
    int a = hs[e], b = hd[e];
    float sa = s[a];
    #pragma unroll
    for (int c = 0; c < 16; ++c)
        atomicAdd(&agg[(long)b * 64 + c0 + c], x[(long)a * 64 + c0 + c] * sa);
}

// ---------------------------------------------------------------------------
// TreeLSTM level kernels
// ---------------------------------------------------------------------------
// Per tree edge: f = sigmoid(h[child] @ Uf^T + Ufb); hsum[par] += h[child];
// csum[par] += f * c[child]
__global__ void tree_edge_kernel(const float* __restrict__ h, const float* __restrict__ c,
                                 const int* __restrict__ tsrc, const int* __restrict__ tdst,
                                 const unsigned short* __restrict__ UfFrag,
                                 const float* __restrict__ Ufb,
                                 float* __restrict__ hsum, float* __restrict__ csum, int Et) {
    __shared__ unsigned short lbf[GW][16][64];
    __shared__ float lh[GW][16][64];
    __shared__ float lc[GW][16][64];
    int lane = threadIdx.x & 31, w = threadIdx.x >> 5;
    int ebase = (blockIdx.x * GW + w) * 16;
    for (int idx = lane; idx < 16 * 64; idx += 32) {
        int rr = idx >> 6, cc = idx & 63;
        int e = ebase + rr;
        float hv = 0.f, cv = 0.f;
        if (e < Et) {
            int sn = tsrc[e];
            hv = h[(long)sn * 64 + cc];
            cv = c[(long)sn * 64 + cc];
        }
        lh[w][rr][cc] = hv; lc[w][rr][cc] = cv; lbf[w][rr][cc] = f2bf(hv);
    }
    __syncthreads();
    if (ebase >= Et) return;
    int r = lane & 15, hi = lane >> 4;
    for (int nt = 0; nt < 4; ++nt) {
        v8f acc = {0.f,0.f,0.f,0.f,0.f,0.f,0.f,0.f};
        for (int ks = 0; ks < 2; ++ks) {
            AFrag a, b;
            const uint4* pa = (const uint4*)&lbf[w][r][ks * 32 + 8 * hi];
            a.q[0] = pa[0]; a.q[1] = pa[2];
            const uint4* pb = (const uint4*)&UfFrag[((nt * 2 + ks) * 32 + lane) * 16];
            b.q[0] = pb[0]; b.q[1] = pb[1];
            acc = wmma_bf16(a.v, b.v, acc);
        }
        int col = nt * 16 + r;
        float bv = Ufb[col];
        #pragma unroll
        for (int v = 0; v < 8; ++v) {
            int rr = v + 8 * hi;
            int e = ebase + rr;
            if (e < Et) {
                float f = sigm(acc[v] + bv);
                int d2 = tdst[e];
                atomicAdd(&hsum[(long)d2 * 64 + col], lh[w][rr][col]);
                atomicAdd(&csum[(long)d2 * 64 + col], f * lc[w][rr][col]);
            }
        }
    }
}
// Per node: iou = iou_x + hsum @ U_iou^T + b_iou; gated update where height==L
__global__ void tree_node_kernel(const float* __restrict__ hsum, const float* __restrict__ csum,
                                 const float* __restrict__ ioux,
                                 const unsigned short* __restrict__ UiouFrag,
                                 const float* __restrict__ biou,
                                 const int* __restrict__ height, int L,
                                 float* __restrict__ h, float* __restrict__ c, int Nn) {
    __shared__ unsigned short lbf[GW][16][64];
    int lane = threadIdx.x & 31, w = threadIdx.x >> 5;
    int tbase = (blockIdx.x * GW + w) * 16;
    for (int idx = lane; idx < 16 * 64; idx += 32) {
        int rr = idx >> 6, cc = idx & 63;
        int row = tbase + rr;
        lbf[w][rr][cc] = f2bf(row < Nn ? hsum[(long)row * 64 + cc] : 0.f);
    }
    __syncthreads();
    if (tbase >= Nn) return;
    int r = lane & 15, hi = lane >> 4;
    for (int kk = 0; kk < 4; ++kk) {        // hcol tile; i/o/u live at ntile kk, kk+4, kk+8
        v8f ai = {0.f,0.f,0.f,0.f,0.f,0.f,0.f,0.f};
        v8f ao = ai, au = ai;
        for (int ks = 0; ks < 2; ++ks) {
            AFrag a, bi, bo, bu;
            const uint4* pa = (const uint4*)&lbf[w][r][ks * 32 + 8 * hi];
            a.q[0] = pa[0]; a.q[1] = pa[2];
            const uint4* pi = (const uint4*)&UiouFrag[(((kk    ) * 2 + ks) * 32 + lane) * 16];
            const uint4* po = (const uint4*)&UiouFrag[(((kk + 4) * 2 + ks) * 32 + lane) * 16];
            const uint4* pu = (const uint4*)&UiouFrag[(((kk + 8) * 2 + ks) * 32 + lane) * 16];
            bi.q[0] = pi[0]; bi.q[1] = pi[1];
            bo.q[0] = po[0]; bo.q[1] = po[1];
            bu.q[0] = pu[0]; bu.q[1] = pu[1];
            ai = wmma_bf16(a.v, bi.v, ai);
            ao = wmma_bf16(a.v, bo.v, ao);
            au = wmma_bf16(a.v, bu.v, au);
        }
        int hcol = kk * 16 + r;
        float bi_ = biou[hcol], bo_ = biou[64 + hcol], bu_ = biou[128 + hcol];
        #pragma unroll
        for (int v = 0; v < 8; ++v) {
            int row = tbase + v + 8 * hi;
            if (row < Nn && height[row] == L) {
                float iv = ai[v] + ioux[(long)row * 192 + hcol]       + bi_;
                float ov = ao[v] + ioux[(long)row * 192 + 64 + hcol]  + bo_;
                float uv = au[v] + ioux[(long)row * 192 + 128 + hcol] + bu_;
                float cs = csum[(long)row * 64 + hcol];
                float cn = sigm(iv) * tanhf(uv) + cs;
                float hn = sigm(ov) * tanhf(cn);
                h[(long)row * 64 + hcol] = hn;
                c[(long)row * 64 + hcol] = cn;
            }
        }
    }
}

// ---------------------------------------------------------------------------
// Pooling + per-graph heads
// acc layout per graph: [0,64) h_call | [64,128) h_host | [128,192) h_tl |
//                       [192,199) ctx | [199] count
// ---------------------------------------------------------------------------
__global__ void pool_kernel(const float* __restrict__ hcall, const float* __restrict__ hhost,
                            const float* __restrict__ htl, const float* __restrict__ ctx,
                            const int* __restrict__ gid, float* __restrict__ acc, int Nn) {
    int n = blockIdx.x * blockDim.x + threadIdx.x;
    if (n >= Nn) return;
    float* a = acc + (long)gid[n] * 200;
    for (int c = 0; c < 64; ++c) {
        atomicAdd(&a[c],       hcall[(long)n * 64 + c]);
        atomicAdd(&a[64 + c],  hhost[(long)n * 64 + c]);
        atomicAdd(&a[128 + c], htl[(long)n * 64 + c]);
    }
    for (int c = 0; c < 7; ++c) atomicAdd(&a[192 + c], ctx[(long)n * 7 + c]);
    atomicAdd(&a[199], 1.f);
}

__global__ void head_kernel(const float* __restrict__ acc,
                            const float* __restrict__ ctxW, const float* __restrict__ ctxb,
                            const float* __restrict__ fuseW, const float* __restrict__ fuseb,
                            const float* __restrict__ hbW, const float* __restrict__ hbb,
                            const float* __restrict__ hc3W, const float* __restrict__ hc3b,
                            const float* __restrict__ htW, const float* __restrict__ htb,
                            float* __restrict__ out, int B) {
    int g = blockIdx.x;
    if (g >= B) return;
    __shared__ float fin[256];
    __shared__ float fused[128];
    const float* a = acc + (long)g * 200;
    float inv = 1.f / fmaxf(a[199], 1.f);
    int t = threadIdx.x;                       // 128 threads
    for (int i = t; i < 192; i += 128) fin[i] = a[i] * inv;
    if (t < 64) {
        float s = ctxb[t];
        for (int j = 0; j < 7; ++j) s += (a[192 + j] * inv) * ctxW[t * 7 + j];
        fin[192 + t] = fmaxf(s, 0.f);
    }
    __syncthreads();
    float s = fuseb[t];
    for (int j = 0; j < 256; ++j) s += fin[j] * fuseW[t * 256 + j];
    fused[t] = fmaxf(s, 0.f);
    __syncthreads();
    if (t < 14) {
        if (t == 0) {
            float r = hbb[0];
            for (int j = 0; j < 128; ++j) r += fused[j] * hbW[j];
            out[g] = r;
        } else if (t < 4) {
            int k = t - 1;
            float r = hc3b[k];
            for (int j = 0; j < 128; ++j) r += fused[j] * hc3W[k * 128 + j];
            out[B + (long)g * 3 + k] = r;
        } else {
            int k = t - 4;
            float r = htb[k];
            for (int j = 0; j < 128; ++j) r += fused[j] * htW[k * 128 + j];
            out[4L * B + (long)g * 10 + k] = r;
        }
    }
}

// ---------------------------------------------------------------------------
// Host orchestration
// ---------------------------------------------------------------------------
static inline unsigned cdiv(long a, long b) { return (unsigned)((a + b - 1) / b); }

extern "C" void kernel_launch(void* const* d_in, const int* in_sizes, int n_in,
                              void* d_out, int out_size, void* d_ws, size_t ws_size,
                              hipStream_t stream) {
    (void)n_in; (void)ws_size;
    const int N  = in_sizes[0];
    const int E  = in_sizes[7];
    const int Eh = in_sizes[9];
    const int Et = in_sizes[11];
    const int B  = out_size / 14;

    const int* api_id    = (const int*)d_in[0];
    const int* status_id = (const int*)d_in[1];
    const int* node_id   = (const int*)d_in[2];
    const int* depth     = (const int*)d_in[3];
    const int* pos       = (const int*)d_in[4];
    const float* lat     = (const float*)d_in[5];
    const float* ctx     = (const float*)d_in[6];
    const int* src       = (const int*)d_in[7];
    const int* dst       = (const int*)d_in[8];
    const int* hsrc      = (const int*)d_in[9];
    const int* hdst      = (const int*)d_in[10];
    const int* tsrc      = (const int*)d_in[11];
    const int* tdst      = (const int*)d_in[12];
    const int* height    = (const int*)d_in[13];
    const int* gid       = (const int*)d_in[14];
    // params in _make_params() insertion order, starting at index 17
    const float* api_emb    = (const float*)d_in[17];
    const float* status_emb = (const float*)d_in[18];
    const float* node_emb   = (const float*)d_in[19];
    const float* depth_emb  = (const float*)d_in[20];
    const float* pos_emb    = (const float*)d_in[21];
    const float* lat_W1     = (const float*)d_in[22];
    const float* lat_b1     = (const float*)d_in[23];
    const float* lat_W2     = (const float*)d_in[24];
    const float* lat_b2     = (const float*)d_in[25];
    const float* merge_W    = (const float*)d_in[26];
    const float* merge_b    = (const float*)d_in[27];
    const float* gcn1_W     = (const float*)d_in[28];
    const float* gcn1_b     = (const float*)d_in[29];
    const float* gcn2_W     = (const float*)d_in[30];
    const float* gcn2_b     = (const float*)d_in[31];
    const float* hgc1_W     = (const float*)d_in[32];
    const float* hgc1_b     = (const float*)d_in[33];
    const float* hgc2_W     = (const float*)d_in[34];
    const float* hgc2_b     = (const float*)d_in[35];
    const float* W_iouf     = (const float*)d_in[36];
    const float* U_iou      = (const float*)d_in[37];
    const float* b_iou      = (const float*)d_in[38];
    const float* Uf_W       = (const float*)d_in[39];
    const float* Uf_b       = (const float*)d_in[40];
    const float* tlout_W    = (const float*)d_in[41];
    const float* tlout_b    = (const float*)d_in[42];
    const float* ctx_W      = (const float*)d_in[43];
    const float* ctx_b      = (const float*)d_in[44];
    const float* fuse_W     = (const float*)d_in[45];
    const float* fuse_b     = (const float*)d_in[46];
    const float* hb_W       = (const float*)d_in[47];
    const float* hb_b       = (const float*)d_in[48];
    const float* hc3_W      = (const float*)d_in[49];
    const float* hc3_b      = (const float*)d_in[50];
    const float* ht_W       = (const float*)d_in[51];
    const float* ht_b       = (const float*)d_in[52];

    // workspace layout (fp32)
    float* ws = (float*)d_ws;
    size_t o = 0;
    float* x      = ws + o; o += (size_t)N * 64;   // features -> later TreeLSTM h
    float* t0     = ws + o; o += (size_t)N * 64;   // scratch -> TreeLSTM c
    float* t1     = ws + o; o += (size_t)N * 64;   // agg scratch / hsum / h_tl
    float* t2     = ws + o; o += (size_t)N * 64;   // agg scratch / csum
    float* h_call = ws + o; o += (size_t)N * 64;
    float* h_host = ws + o; o += (size_t)N * 64;
    float* iou_x  = ws + o; o += (size_t)N * 192;
    float* degc   = ws + o; o += (size_t)N;
    float* dho    = ws + o; o += (size_t)N;
    float* dhi    = ws + o; o += (size_t)N;
    float* acc    = ws + o; o += (size_t)B * 200;
    unsigned short* fr = (unsigned short*)(ws + o);
    unsigned short* fMerge = fr;            // 64x192
    unsigned short* fG1    = fr + 12288;    // 64x64
    unsigned short* fG2    = fG1 + 4096;
    unsigned short* fH1    = fG2 + 4096;
    unsigned short* fH2    = fH1 + 4096;
    unsigned short* fWiou  = fH2 + 4096;    // 192x64 (first 192 rows of W_iouf)
    unsigned short* fUiou  = fWiou + 12288; // 192x64
    unsigned short* fUf    = fUiou + 12288; // 64x64
    unsigned short* fTl    = fUf + 4096;    // 64x64

    const long NH = (long)N * 64;
    const unsigned tilesN = cdiv(cdiv(N, 16), GW);
    const unsigned tilesE = cdiv(cdiv(Et, 16), GW);

    // 1) weight fragmentization (cheap; redone each call -> deterministic)
    wfrag_kernel<<<cdiv(64 * 192, 256), 256, 0, stream>>>(merge_W, fMerge, 64, 192);
    wfrag_kernel<<<cdiv(64 * 64, 256), 256, 0, stream>>>(gcn1_W, fG1, 64, 64);
    wfrag_kernel<<<cdiv(64 * 64, 256), 256, 0, stream>>>(gcn2_W, fG2, 64, 64);
    wfrag_kernel<<<cdiv(64 * 64, 256), 256, 0, stream>>>(hgc1_W, fH1, 64, 64);
    wfrag_kernel<<<cdiv(64 * 64, 256), 256, 0, stream>>>(hgc2_W, fH2, 64, 64);
    wfrag_kernel<<<cdiv(192 * 64, 256), 256, 0, stream>>>(W_iouf, fWiou, 192, 64);
    wfrag_kernel<<<cdiv(192 * 64, 256), 256, 0, stream>>>(U_iou, fUiou, 192, 64);
    wfrag_kernel<<<cdiv(64 * 64, 256), 256, 0, stream>>>(Uf_W, fUf, 64, 64);
    wfrag_kernel<<<cdiv(64 * 64, 256), 256, 0, stream>>>(tlout_W, fTl, 64, 64);

    // 2) embeddings + lat MLP + merge -> x[N,64]
    embed_merge_kernel<<<tilesN, 128, 0, stream>>>(api_id, status_id, node_id, depth, pos, lat,
        api_emb, status_emb, node_emb, depth_emb, pos_emb,
        lat_W1, lat_b1, lat_W2, lat_b2, fMerge, merge_b, x, N);

    // 3) iou_x = x @ W_iouf[:192]^T
    gemm_wmma_kernel<<<tilesN, 128, 0, stream>>>(x, nullptr, fWiou, nullptr, iou_x, N, 64, 192, 0);

    // 4) degrees (call graph symmetrized => dout==din==degc; self loop = init 1)
    fill_kernel<<<cdiv(N, 256), 256, 0, stream>>>(degc, 1.f, N);
    fill_kernel<<<cdiv(N, 256), 256, 0, stream>>>(dho, 0.f, N);
    fill_kernel<<<cdiv(N, 256), 256, 0, stream>>>(dhi, 0.f, N);
    call_deg_kernel<<<cdiv(E, 256), 256, 0, stream>>>(src, dst, degc, E);
    host_deg_kernel<<<cdiv(Eh, 256), 256, 0, stream>>>(hsrc, hdst, dho, dhi, Eh);
    rsqrt_kernel<<<cdiv(N, 256), 256, 0, stream>>>(degc, N);
    rsqrt_kernel<<<cdiv(N, 256), 256, 0, stream>>>(dho, N);
    rsqrt_kernel<<<cdiv(N, 256), 256, 0, stream>>>(dhi, N);

    // 5) call-graph GCN x2
    initagg_kernel<<<cdiv(NH, 256), 256, 0, stream>>>(x, degc, t1, N);
    scatter_call_kernel<<<cdiv(4L * E, 256), 256, 0, stream>>>(x, degc, src, dst, t1, E);
    gemm_wmma_kernel<<<tilesN, 128, 0, stream>>>(t1, degc, fG1, gcn1_b, t1, N, 64, 64, 2);
    initagg_kernel<<<cdiv(NH, 256), 256, 0, stream>>>(t1, degc, t2, N);
    scatter_call_kernel<<<cdiv(4L * E, 256), 256, 0, stream>>>(t1, degc, src, dst, t2, E);
    gemm_wmma_kernel<<<tilesN, 128, 0, stream>>>(t2, degc, fG2, gcn2_b, h_call, N, 64, 64, 0);

    // 6) host-graph GCN x2 (edge list already contains both directions + loops)
    fill_kernel<<<cdiv(NH, 256), 256, 0, stream>>>(t1, 0.f, NH);
    scatter_host_kernel<<<cdiv(4L * Eh, 256), 256, 0, stream>>>(x, dho, hsrc, hdst, t1, Eh);
    gemm_wmma_kernel<<<tilesN, 128, 0, stream>>>(t1, dhi, fH1, hgc1_b, t1, N, 64, 64, 2);
    fill_kernel<<<cdiv(NH, 256), 256, 0, stream>>>(t2, 0.f, NH);
    scatter_host_kernel<<<cdiv(4L * Eh, 256), 256, 0, stream>>>(t1, dho, hsrc, hdst, t2, Eh);
    gemm_wmma_kernel<<<tilesN, 128, 0, stream>>>(t2, dhi, fH2, hgc2_b, h_host, N, 64, 64, 0);

    // 7) ChildSum TreeLSTM, level-synchronous (h reuses x buffer, c reuses t0)
    fill_kernel<<<cdiv(NH, 256), 256, 0, stream>>>(x, 0.f, NH);
    fill_kernel<<<cdiv(NH, 256), 256, 0, stream>>>(t0, 0.f, NH);
    for (int L = 0; L < LEVELS; ++L) {
        fill_kernel<<<cdiv(NH, 256), 256, 0, stream>>>(t1, 0.f, NH);   // hsum
        fill_kernel<<<cdiv(NH, 256), 256, 0, stream>>>(t2, 0.f, NH);   // csum
        tree_edge_kernel<<<tilesE, 128, 0, stream>>>(x, t0, tsrc, tdst, fUf, Uf_b, t1, t2, Et);
        tree_node_kernel<<<tilesN, 128, 0, stream>>>(t1, t2, iou_x, fUiou, b_iou,
                                                     height, L, x, t0, N);
    }
    // h_tl = relu(h) @ tlout^T + b  -> t1
    gemm_wmma_kernel<<<tilesN, 128, 0, stream>>>(x, nullptr, fTl, tlout_b, t1, N, 64, 64, 1);

    // 8) graph pooling + heads
    fill_kernel<<<cdiv((long)B * 200, 256), 256, 0, stream>>>(acc, 0.f, (long)B * 200);
    pool_kernel<<<cdiv(N, 256), 256, 0, stream>>>(h_call, h_host, t1, ctx, gid, acc, N);
    head_kernel<<<(unsigned)B, 128, 0, stream>>>(acc, ctx_W, ctx_b, fuse_W, fuse_b,
                                                 hb_W, hb_b, hc3_W, hc3_b, ht_W, ht_b,
                                                 (float*)d_out, B);
}
